// PriorLoss_53738630807685
// MI455X (gfx1250) — matrix operand order repaired and checked by reading, not compiled, not run on hardware
//
#include <hip/hip_runtime.h>
#include <hip/hip_bf16.h>
#include <math.h>

// ---------------------------------------------------------------------------
// MI455X / gfx1250 implementation.
// Dominant cost: 4096x16384 cosine GEMM with K=768 (~103 GFLOP) -> f16 WMMA,
// fp32 accumulate, fused argmin. A fragments are register-resident per wave
// (24 x v16h = 192 VGPRs, enabled by __launch_bounds__(256)); only B streams
// from L2 (24MB operand << 192MB L2). Convs (24 GFLOP) -> WMMA implicit GEMM
// (9 shifted GEMMs over K=Cin, HWC activations, fragment-swizzled weights).
// ---------------------------------------------------------------------------

typedef __attribute__((ext_vector_type(16))) _Float16 v16h;
typedef __attribute__((ext_vector_type(8)))  float    v8f;

union F16x16 {
    v16h v;
    _Float16 h[16];
    float4 f4[2];
};

#define EPS 1e-8f

// ---------------------------------------------------------------------------
// 1) bilinear 0.5x downsample (== 2x2 average) + ImageNet normalize
// ---------------------------------------------------------------------------
__global__ __launch_bounds__(256) void k_downsample(const float* __restrict__ img,
                                                    float* __restrict__ xin) {
    int i = blockIdx.x * blockDim.x + threadIdx.x;
    if (i >= 256 * 256) return;
    int h = i >> 8, w = i & 255;
    const float mean[3] = {0.485f, 0.456f, 0.406f};
    const float stdv[3] = {0.229f, 0.224f, 0.225f};
    for (int c = 0; c < 3; ++c) {
        const float* p = img + (size_t)c * 512 * 512;
        float v = 0.25f * (p[(2 * h) * 512 + 2 * w] + p[(2 * h) * 512 + 2 * w + 1] +
                           p[(2 * h + 1) * 512 + 2 * w] + p[(2 * h + 1) * 512 + 2 * w + 1]);
        xin[(size_t)i * 3 + c] = (v - mean[c]) / stdv[c];
    }
}

// ---------------------------------------------------------------------------
// 2) conv0: 3->64 at 256x256, direct (K=27 too small for WMMA). Output f16 HWC.
// ---------------------------------------------------------------------------
__global__ __launch_bounds__(64) void k_conv0(const float* __restrict__ xin,   // [256][256][3]
                                              const float* __restrict__ w,     // [64][3][3][3]
                                              const float* __restrict__ b,     // [64]
                                              _Float16* __restrict__ out) {    // [256][256][64]
    int x = blockIdx.x * blockDim.x + threadIdx.x;
    int y = blockIdx.y;
    if (x >= 256) return;
    float acc[64];
#pragma unroll
    for (int co = 0; co < 64; ++co) acc[co] = b[co];
    for (int kh = -1; kh <= 1; ++kh) {
        int hh = y + kh;
        if (hh < 0 || hh >= 256) continue;
        for (int kw = -1; kw <= 1; ++kw) {
            int ww = x + kw;
            if (ww < 0 || ww >= 256) continue;
            const float* px = xin + ((size_t)hh * 256 + ww) * 3;
            float v0 = px[0], v1 = px[1], v2 = px[2];
            int tap = (kh + 1) * 3 + (kw + 1);
            const float* wp = w + tap;
#pragma unroll
            for (int co = 0; co < 64; ++co) {
                const float* q = wp + (size_t)co * 27;
                acc[co] += v0 * q[0] + v1 * q[9] + v2 * q[18];
            }
        }
    }
    _Float16* op = out + ((size_t)y * 256 + x) * 64;
#pragma unroll
    for (int co = 0; co < 64; ++co) op[co] = (_Float16)fmaxf(acc[co], 0.f);
}

// ---------------------------------------------------------------------------
// 3) weight repack into WMMA-B-fragment-linear order:
//    fp32 OIHW [COUT][CIN][3][3] -> f16 [9][CIN/32][COUT/16][n(16)][hi(2)][16]
//    so a conv B fragment is one contiguous 32B chunk per lane.
// ---------------------------------------------------------------------------
__global__ __launch_bounds__(256) void k_wprep(const float* __restrict__ w, int CIN, int COUT,
                                               _Float16* __restrict__ wt) {
    size_t t = blockIdx.x * (size_t)blockDim.x + threadIdx.x;
    size_t total = (size_t)COUT * CIN * 9;
    if (t >= total) return;
    int tap = (int)(t % 9);
    size_t r = t / 9;
    int ci = (int)(r % CIN);
    int co = (int)(r / CIN);
    // B fragment element: lane (n,hi), elem j holds W[K = kcb*32 + hi*16 + j][co = cobt*16 + n]
    int kcb = ci >> 5;
    int rem = ci & 31;
    int hi  = rem >> 4;
    int j   = rem & 15;
    int cobt = co >> 4;
    int n    = co & 15;
    size_t dst = ((((size_t)(tap * (CIN / 32) + kcb) * (COUT / 16) + cobt) * 16 + n) * 2 + hi) * 16 + j;
    wt[dst] = (_Float16)w[((size_t)co * CIN + ci) * 9 + tap];
}

// ---------------------------------------------------------------------------
// 4) WMMA implicit-GEMM 3x3 conv, stride 1, pad 1, fused bias+ReLU.
//    One wave = one [16 pixels x 16 out-ch] tile; 9 taps x (CIN/32) WMMAs.
//    A 16x32 / B 32x16 use contiguous k = hi*16 + j (same in both); f32 C/D per CDNA5 ISA 7.12.2.
// ---------------------------------------------------------------------------
template <int CIN, int COUT>
__global__ __launch_bounds__(32) void k_conv_wmma(
        const _Float16* __restrict__ in,   // [H][W][CIN]
        const _Float16* __restrict__ wt,   // fragment-swizzled, see k_wprep
        const float* __restrict__ bias,    // [COUT]
        _Float16* __restrict__ out_h,      // [H][W][COUT] or null
        float* __restrict__ out_f,         // [COUT][H*W] or null
        int H, int W) {
    const int lane = threadIdx.x;
    const int n  = lane & 15;   // WMMA column (out channel)
    const int hi = lane >> 4;   // half-wave selector
    const int w0  = blockIdx.x * 16;
    const int h   = blockIdx.y;
    const int cobt = blockIdx.z;        // COUT tile index

    v8f acc;
    float bv = bias[cobt * 16 + n];
#pragma unroll
    for (int r = 0; r < 8; ++r) acc[r] = bv;

    for (int tap = 0; tap < 9; ++tap) {
        int kh = tap / 3 - 1, kw = tap % 3 - 1;
        int hh = h + kh;
        if (hh < 0 || hh >= H) continue;                  // uniform per block
        const _Float16* inrow = in + (size_t)hh * W * CIN;
        int wcol = w0 + n + kw;                           // A row M = n
        bool inb = (wcol >= 0) && (wcol < W);
        const _Float16* ap = inrow + (size_t)(inb ? wcol : 0) * CIN;
#pragma unroll
        for (int kc = 0; kc < CIN; kc += 32) {
            int kcb = kc >> 5;
            F16x16 a;
            if (inb) {
                // lanes<16: K = kc+0..15 ; lanes>=16: K = kc+16..31
                const float4* ap4 = (const float4*)(ap + kc + hi * 16);
                a.f4[0] = ap4[0];
                a.f4[1] = ap4[1];
            } else {
#pragma unroll
                for (int j = 0; j < 16; ++j) a.h[j] = (_Float16)0.f;
            }
            F16x16 bfr;   // contiguous 32B fragment chunk for this lane
            const float4* bp4 = (const float4*)(wt +
                ((((size_t)(tap * (CIN / 32) + kcb) * (COUT / 16) + cobt) * 16 + n) * 2 + hi) * 16);
            bfr.f4[0] = bp4[0];
            bfr.f4[1] = bp4[1];
            acc = __builtin_amdgcn_wmma_f32_16x16x32_f16(
                false, a.v, false, bfr.v, (short)0, acc, false, false);
        }
    }
#pragma unroll
    for (int r = 0; r < 8; ++r) {
        int m = r + hi * 8;                 // pixel within strip
        float v = fmaxf(acc[r], 0.f);       // fused ReLU
        int x = w0 + m;
        if (out_h) out_h[((size_t)h * W + x) * COUT + cobt * 16 + n] = (_Float16)v;
        if (out_f) out_f[(size_t)(cobt * 16 + n) * ((size_t)H * W) + (size_t)h * W + x] = v;
    }
}

// ---------------------------------------------------------------------------
// 5) 2x2 maxpool, f16 HWC
// ---------------------------------------------------------------------------
template <int C>
__global__ __launch_bounds__(256) void k_pool(const _Float16* __restrict__ in, int Ho, int Wo,
                                              _Float16* __restrict__ out) {
    size_t t = blockIdx.x * (size_t)blockDim.x + threadIdx.x;
    size_t total = (size_t)Ho * Wo * C;
    if (t >= total) return;
    int c = (int)(t % C);
    size_t p = t / C;
    int w = (int)(p % Wo);
    int h = (int)(p / Wo);
    int Wi = Wo * 2;
    const _Float16* r0 = in + ((size_t)(2 * h) * Wi + 2 * w) * C + c;
    const _Float16* r1 = r0 + (size_t)Wi * C;
    float m = fmaxf(fmaxf((float)r0[0], (float)r0[C]), fmaxf((float)r1[0], (float)r1[C]));
    out[t] = (_Float16)m;
}

// ---------------------------------------------------------------------------
// 6) column-normalize (over C) a c-major fp32 matrix [C][N] into row-major
//    f16 [N][C] (the WMMA-friendly layout). Coalesced reads over N.
// ---------------------------------------------------------------------------
__global__ __launch_bounds__(256) void k_colnorm(const float* __restrict__ X, int C, int N,
                                                 _Float16* __restrict__ Y) {
    int q = blockIdx.x * blockDim.x + threadIdx.x;
    if (q >= N) return;
    float s = 0.f;
    for (int c = 0; c < C; ++c) {
        float v = X[(size_t)c * N + q];
        s += v * v;
    }
    float inv = 1.f / (sqrtf(s) + EPS);
    for (int c = 0; c < C; ++c)
        Y[(size_t)q * C + c] = (_Float16)(X[(size_t)c * N + q] * inv);
}

// ---------------------------------------------------------------------------
// 7) Fused cosine GEMM + L1 term + row argmin.
//    Grid: 256 q-tiles of 16. Block: 8 waves. Each wave keeps the full A tile
//    (24 x v16h fragments = 192 VGPRs) register-resident and streams B for its
//    128 k-tiles (2 contiguous b128 loads per WMMA), then reduces (min,idx)
//    across lanes/waves with first-index tie-break.
// ---------------------------------------------------------------------------
__global__ __launch_bounds__(256) void k_argmin(
        const _Float16* __restrict__ anh,  // [4096][768]
        const _Float16* __restrict__ bnh,  // [16384][768]
        const float* __restrict__ pMat,    // [4096]
        const float* __restrict__ Mat,     // [16384]
        int* __restrict__ idx) {
    __shared__ float redMin[16][8];
    __shared__ int   redIdx[16][8];

    const int lane = threadIdx.x & 31;
    const int wave = threadIdx.x >> 5;
    const int n  = lane & 15;
    const int hi = lane >> 4;
    const int qb = blockIdx.x * 16;

    // Register-resident A fragments for this q-tile (row M = n for this lane).
    v16h afrag[24];
    {
        const _Float16* arow = anh + (size_t)(qb + n) * 768;
#pragma unroll
        for (int kcb = 0; kcb < 24; ++kcb) {
            F16x16 a;
            const float4* ap4 = (const float4*)(arow + kcb * 32 + hi * 16);
            a.f4[0] = ap4[0];   // K = kc+0..7   (lanes<16) / kc+16..23 (lanes>=16)
            a.f4[1] = ap4[1];   // K = kc+8..15  (lanes<16) / kc+24..31 (lanes>=16)
            afrag[kcb] = a.v;
        }
    }

    float pq[8];
#pragma unroll
    for (int r = 0; r < 8; ++r) pq[r] = pMat[qb + r + hi * 8];

    float best[8];
    int bidx[8];
#pragma unroll
    for (int r = 0; r < 8; ++r) { best[r] = 3.4e38f; bidx[r] = 0; }

    for (int kt = wave; kt < 1024; kt += 8) {
        int kb = kt * 16;
        // next-tile prefetch for this wave's B stream (near caches)
        if (kt + 8 < 1024)
            __builtin_prefetch(bnh + (size_t)(kb + 128 + n) * 768, 0, 3);
        float mk = Mat[kb + n];
        v8f acc;
#pragma unroll
        for (int r = 0; r < 8; ++r) acc[r] = 0.f;
        const _Float16* brow = bnh + (size_t)(kb + n) * 768;  // B column N = n
#pragma unroll
        for (int kcb = 0; kcb < 24; ++kcb) {
            F16x16 b;   // lanes<16: K = kc+0..15 ; lanes>=16: K = kc+16..31
            const float4* bp4 = (const float4*)(brow + kcb * 32 + hi * 16);
            b.f4[0] = bp4[0];
            b.f4[1] = bp4[1];
            acc = __builtin_amdgcn_wmma_f32_16x16x32_f16(
                false, afrag[kcb], false, b.v, (short)0, acc, false, false);
        }
        int k = kb + n;
#pragma unroll
        for (int r = 0; r < 8; ++r) {
            float cost = 1.f - acc[r] + fabsf(pq[r] - mk);
            if (cost < best[r]) { best[r] = cost; bidx[r] = k; }
            else if (cost == best[r] && k < bidx[r]) bidx[r] = k;
        }
    }

    // cross-lane argmin within each 16-lane half (same q rows, different k cols)
#pragma unroll
    for (int r = 0; r < 8; ++r) {
        for (int off = 8; off; off >>= 1) {
            float ob = __shfl_xor(best[r], off, 16);
            int   oi = __shfl_xor(bidx[r], off, 16);
            if (ob < best[r] || (ob == best[r] && oi < bidx[r])) {
                best[r] = ob; bidx[r] = oi;
            }
        }
    }
    if (n == 0) {
#pragma unroll
        for (int r = 0; r < 8; ++r) {
            redMin[r + hi * 8][wave] = best[r];
            redIdx[r + hi * 8][wave] = bidx[r];
        }
    }
    __syncthreads();
    if (threadIdx.x < 16) {
        float bb = redMin[threadIdx.x][0];
        int bi = redIdx[threadIdx.x][0];
        for (int wv = 1; wv < 8; ++wv) {
            float ob = redMin[threadIdx.x][wv];
            int oi = redIdx[threadIdx.x][wv];
            if (ob < bb || (ob == bb && oi < bi)) { bb = ob; bi = oi; }
        }
        idx[qb + threadIdx.x] = bi;
    }
}

// ---------------------------------------------------------------------------
// 8) gather + cos_loss over the height axis; deterministic tree reduction.
//    loss = mean over (c,w) of 1 - sum_h(t*a) / ((||t||+eps)(||a||+eps))
// ---------------------------------------------------------------------------
__global__ __launch_bounds__(256) void k_loss_partial(
        const float* __restrict__ F,  // [768][4096]
        const float* __restrict__ B,  // [768][16384]
        const int* __restrict__ idx,  // [4096]
        float* __restrict__ partial) {
    int t = blockIdx.x * blockDim.x + threadIdx.x;
    float val = 0.f;
    if (t < 768 * 64) {
        int c = t >> 6, w = t & 63;
        const float* a = F + (size_t)c * 4096;
        const float* brow = B + (size_t)c * 16384;
        float aa = 0.f, tt = 0.f, ta = 0.f;
        for (int h = 0; h < 64; ++h) {
            int q = h * 64 + w;
            float av = a[q];
            float tv = brow[idx[q]];
            aa += av * av; tt += tv * tv; ta += av * tv;
        }
        val = 1.f - ta / ((sqrtf(tt) + EPS) * (sqrtf(aa) + EPS));
    }
    __shared__ float s[256];
    s[threadIdx.x] = val;
    __syncthreads();
    for (int o = 128; o; o >>= 1) {
        if (threadIdx.x < o) s[threadIdx.x] += s[threadIdx.x + o];
        __syncthreads();
    }
    if (threadIdx.x == 0) partial[blockIdx.x] = s[0];
}

__global__ __launch_bounds__(256) void k_loss_final(const float* __restrict__ partial,
                                                    int np, float* out) {
    __shared__ float s[256];
    float v = 0.f;
    for (int i = threadIdx.x; i < np; i += 256) v += partial[i];
    s[threadIdx.x] = v;
    __syncthreads();
    for (int o = 128; o; o >>= 1) {
        if (threadIdx.x < o) s[threadIdx.x] += s[threadIdx.x + o];
        __syncthreads();
    }
    if (threadIdx.x == 0) out[0] = s[0] / (768.f * 64.f);
}

// ---------------------------------------------------------------------------
extern "C" void kernel_launch(void* const* d_in, const int* in_sizes, int n_in,
                              void* d_out, int out_size, void* d_ws, size_t ws_size,
                              hipStream_t stream) {
    (void)in_sizes; (void)n_in; (void)out_size; (void)ws_size;
    const float* img   = (const float*)d_in[0];   // [3,512,512]
    const float* sfeat = (const float*)d_in[1];   // [768,16384]
    const float* Mat   = (const float*)d_in[2];   // [16384]
    const float* pMat  = (const float*)d_in[3];   // [4096]
    const float* w0  = (const float*)d_in[4];  const float* b0  = (const float*)d_in[5];
    const float* w2  = (const float*)d_in[6];  const float* b2  = (const float*)d_in[7];
    const float* w5  = (const float*)d_in[8];  const float* b5  = (const float*)d_in[9];
    const float* w7  = (const float*)d_in[10]; const float* b7  = (const float*)d_in[11];
    const float* w10 = (const float*)d_in[12]; const float* b10 = (const float*)d_in[13];
    const float* w12 = (const float*)d_in[14]; const float* b12 = (const float*)d_in[15];
    const float* w14 = (const float*)d_in[16]; const float* b14 = (const float*)d_in[17];
    float* out = (float*)d_out;

    size_t off = 0;
    auto carve = [&](size_t bytes) {
        void* p = (char*)d_ws + off;
        off += (bytes + 255) & ~(size_t)255;
        return p;
    };
    float*    xin  = (float*)carve(256 * 256 * 3 * 4);
    _Float16* bufA = (_Float16*)carve((size_t)256 * 256 * 64 * 2);   // 8 MB ping
    _Float16* bufB = (_Float16*)carve((size_t)256 * 256 * 64 * 2);   // 8 MB pong
    float*    F    = (float*)carve((size_t)768 * 4096 * 4);          // f1|f2|f3, c-major
    _Float16* anh  = (_Float16*)carve((size_t)4096 * 768 * 2);
    _Float16* bnh  = (_Float16*)carve((size_t)16384 * 768 * 2);
    _Float16* wt2  = (_Float16*)carve((size_t)9 * 64 * 64 * 2);
    _Float16* wt5  = (_Float16*)carve((size_t)9 * 64 * 128 * 2);
    _Float16* wt7  = (_Float16*)carve((size_t)9 * 128 * 128 * 2);
    _Float16* wt10 = (_Float16*)carve((size_t)9 * 128 * 256 * 2);
    _Float16* wt12 = (_Float16*)carve((size_t)9 * 256 * 256 * 2);
    _Float16* wt14 = (_Float16*)carve((size_t)9 * 256 * 256 * 2);
    int*   idx      = (int*)carve(4096 * 4);
    float* partials = (float*)carve(256 * 4);

    // --- preprocessing ---
    k_downsample<<<256, 256, 0, stream>>>(img, xin);
    k_wprep<<<((9 * 64 * 64) + 255) / 256, 256, 0, stream>>>(w2, 64, 64, wt2);
    k_wprep<<<((9 * 64 * 128) + 255) / 256, 256, 0, stream>>>(w5, 64, 128, wt5);
    k_wprep<<<((9 * 128 * 128) + 255) / 256, 256, 0, stream>>>(w7, 128, 128, wt7);
    k_wprep<<<((9 * 128 * 256) + 255) / 256, 256, 0, stream>>>(w10, 128, 256, wt10);
    k_wprep<<<((9 * 256 * 256) + 255) / 256, 256, 0, stream>>>(w12, 256, 256, wt12);
    k_wprep<<<((9 * 256 * 256) + 255) / 256, 256, 0, stream>>>(w14, 256, 256, wt14);

    // --- VGG trunk ---
    k_conv0<<<dim3(4, 256), 64, 0, stream>>>(xin, w0, b0, bufA);                 // 3->64 @256
    k_conv_wmma<64, 64><<<dim3(16, 256, 4), 32, 0, stream>>>(bufA, wt2, b2,
                                                             bufB, nullptr, 256, 256);
    k_pool<64><<<(128 * 128 * 64) / 256, 256, 0, stream>>>(bufB, 128, 128, bufA);
    k_conv_wmma<64, 128><<<dim3(8, 128, 8), 32, 0, stream>>>(bufA, wt5, b5,
                                                             bufB, nullptr, 128, 128);
    k_conv_wmma<128, 128><<<dim3(8, 128, 8), 32, 0, stream>>>(bufB, wt7, b7,
                                                              bufA, nullptr, 128, 128);
    k_pool<128><<<(64 * 64 * 128) / 256, 256, 0, stream>>>(bufA, 64, 64, bufB);
    k_conv_wmma<128, 256><<<dim3(4, 64, 16), 32, 0, stream>>>(bufB, wt10, b10,
                                                              bufA, F + 0, 64, 64);      // f1
    k_conv_wmma<256, 256><<<dim3(4, 64, 16), 32, 0, stream>>>(bufA, wt12, b12,
                                                              bufB, F + (size_t)256 * 4096, 64, 64); // f2
    k_conv_wmma<256, 256><<<dim3(4, 64, 16), 32, 0, stream>>>(bufB, wt14, b14,
                                                              nullptr, F + (size_t)512 * 4096, 64, 64); // f3

    // --- cosine-normalized operands (row-major f16) ---
    k_colnorm<<<16, 256, 0, stream>>>(F, 768, 4096, anh);
    k_colnorm<<<64, 256, 0, stream>>>(sfeat, 768, 16384, bnh);

    // --- fused GEMM + argmin (the 103 GFLOP kernel) ---
    k_argmin<<<256, 256, 0, stream>>>(anh, bnh, pMat, Mat, idx);

    // --- gather + cos loss ---
    k_loss_partial<<<192, 256, 0, stream>>>(F, sfeat, idx, partials);
    k_loss_final<<<1, 256, 0, stream>>>(partials, 192, out);
}
